// DeltaAttention_11063835755022
// MI455X (gfx1250) — matrix operand order repaired and checked by reading, not compiled, not run on hardware
//
#include <hip/hip_runtime.h>
#include <hip/hip_bf16.h>
#include <math.h>

// Problem constants (B,S,D,H from reference)
#define BB   2
#define SS   2048
#define DI   1024
#define HH   16
#define HDIM 64
#define MROWS (BB * SS)   // 4096

typedef unsigned short u16;
typedef __attribute__((ext_vector_type(4)))  unsigned short u16x4;
typedef __attribute__((ext_vector_type(8)))  unsigned short u16x8;
typedef __attribute__((ext_vector_type(16))) unsigned short u16x16;
typedef __attribute__((ext_vector_type(16))) __bf16         v16bf;
typedef __attribute__((ext_vector_type(8)))  float          v8f;
typedef __attribute__((ext_vector_type(4)))  unsigned int   ui32x4;
typedef __attribute__((ext_vector_type(8)))  int            i32x8;
typedef __attribute__((ext_vector_type(4)))  int            i32x4;

union FragU { u16x16 v; u16x8 h[2]; };

static __device__ __forceinline__ u16 f2bf(float f) {
    unsigned u = __float_as_uint(f);
    unsigned r = u + 0x7FFFu + ((u >> 16) & 1u);   // RNE
    return (u16)(r >> 16);
}

static __device__ __forceinline__ v8f wmma_bf16(u16x16 a, u16x16 b, v8f c) {
    return __builtin_amdgcn_wmma_f32_16x16x32_bf16(
        /*neg_a=*/false, __builtin_bit_cast(v16bf, a),
        /*neg_b=*/false, __builtin_bit_cast(v16bf, b),
        /*c_mod=*/(short)0, c, /*reuse_a=*/false, /*reuse_b=*/false);
}

#if __has_builtin(__builtin_amdgcn_tensor_load_to_lds)
#define HAVE_TDM 1
#else
#define HAVE_TDM 0
#endif

// Issue a 1-D TDM copy of `n_elem` bf16 elements from `gaddr` into LDS at
// byte offset `lds_off`. D# per CDNA5 ISA ch.8: group0 (count=1, lds_addr,
// global_addr, type=2), group1 (data_size=1 -> 2B, tensor_dim0 big enough,
// tile_dim0 = #elements, tile_dim1=0 -> 1D).
static __device__ __forceinline__ void tdm_load_1d(unsigned long long gaddr,
                                                   unsigned lds_off,
                                                   unsigned n_elem) {
#if HAVE_TDM
    ui32x4 g0;
    g0[0] = 1u;                                        // count=1, user descriptor
    g0[1] = lds_off;                                   // lds_addr (bytes)
    g0[2] = (unsigned)(gaddr & 0xFFFFFFFFu);           // global_addr[31:0]
    g0[3] = (unsigned)((gaddr >> 32) & 0x01FFFFFFu)    // global_addr[56:32]
          | (2u << 30);                                // type=2 ("image")
    i32x8 g1;
    g1[0] = (int)(1u << 16);                           // data_size=1 (2 bytes)
    // tensor_dim0 (bits 79:48) = MROWS*DI (no OOB clipping)
    const unsigned td0 = (unsigned)MROWS * (unsigned)DI;
    g1[1] = (int)((td0 & 0xFFFFu) << 16);              // tensor_dim0[15:0]
    g1[2] = (int)(((td0 >> 16) & 0xFFFFu) | (1u << 16)); // dim0[31:16], tensor_dim1=1
    g1[3] = (int)((n_elem & 0xFFFFu) << 16);           // tile_dim0 (<=65535)
    g1[4] = 0;                                         // tile_dim1=0, tile_dim2=0
    g1[5] = (int)(DI);                                 // tensor_dim0_stride[31:0]
    g1[6] = 0;
    g1[7] = 0;
    i32x4 gz = {0, 0, 0, 0};
#if defined(__clang_major__) && (__clang_major__ >= 23)
    i32x8 gz8 = {0, 0, 0, 0, 0, 0, 0, 0};
    __builtin_amdgcn_tensor_load_to_lds(g0, g1, gz, gz, gz8, 0);
#else
    __builtin_amdgcn_tensor_load_to_lds(g0, g1, gz, gz, 0);
#endif
#endif
}

static __device__ __forceinline__ void wait_tensorcnt0() {
#if __has_builtin(__builtin_amdgcn_s_wait_tensorcnt)
    __builtin_amdgcn_s_wait_tensorcnt(0);
#else
    asm volatile("s_wait_tensorcnt 0" ::: "memory");
#endif
}

// ---------------------------------------------------------------- pack f32->bf16
__global__ __launch_bounds__(256) void k_pack_bf16(const float* __restrict__ in,
                                                   u16* __restrict__ out) {
    int i = blockIdx.x * 256 + threadIdx.x;      // one float4 per thread
    float4 f = ((const float4*)in)[i];
    u16x4 o;
    o[0] = f2bf(f.x); o[1] = f2bf(f.y); o[2] = f2bf(f.z); o[3] = f2bf(f.w);
    ((u16x4*)out)[i] = o;
}

// ------------------------------------------- transpose + pack DxD weight -> bf16 W^T
__global__ __launch_bounds__(256) void k_packT_bf16(const float* __restrict__ W,
                                                    u16* __restrict__ WT) {
    int id = blockIdx.x * 256 + threadIdx.x;     // over DI*DI
    int n = id / DI;
    int k = id - n * DI;
    WT[id] = f2bf(W[(size_t)k * DI + n]);
}

// ---------------------------------------------------------------- WMMA GEMM
// C[M,1024] = A[M,1024](bf16) * WT[1024,1024](bf16, row=N) + bias
// grid = (DI/64, M/16), block = 128 (4 waves, each one 16x16 C tile).
// The 16x1024 A tile (contiguous 32KB) is staged to LDS once per block via the
// Tensor Data Mover and reused by all 4 waves.
__global__ __launch_bounds__(128) void k_gemm_bf16(const u16* __restrict__ A,
                                                   const u16* __restrict__ WT,
                                                   const float* __restrict__ bias,
                                                   float* __restrict__ C) {
    __shared__ u16 ldsA[16 * DI] __attribute__((aligned(64)));   // 32 KB

    const int lane = threadIdx.x & 31;
    const int wave = threadIdx.x >> 5;
    const int hg   = lane >> 4;
    const int l16  = lane & 15;
    const int tileN = blockIdx.x * 64 + wave * 16;
    const int tileM = blockIdx.y * 16;

#if HAVE_TDM
    if (wave == 0) {
        unsigned long long ga =
            (unsigned long long)(const void*)(A + (size_t)tileM * DI);
        unsigned lds_off = (unsigned)(size_t)(void*)&ldsA[0];
        tdm_load_1d(ga, lds_off, 16 * DI);
        wait_tensorcnt0();
    }
#else
    // Fallback: cooperative copy (tile is contiguous)
    {
        const u16x16* src = (const u16x16*)(A + (size_t)tileM * DI);
        u16x16* dst = (u16x16*)ldsA;
        for (int i = threadIdx.x; i < 16 * DI / 16; i += 128) dst[i] = src[i];
    }
#endif
    __syncthreads();

    const u16* arow = ldsA + (size_t)l16 * DI;
    const u16* brow = WT + (size_t)(tileN + l16) * DI;

    v8f acc = {0.f, 0.f, 0.f, 0.f, 0.f, 0.f, 0.f, 0.f};

    for (int k = 0; k < DI; k += 32) {
        FragU af;
        af.h[0] = *(const u16x8*)(arow + k + hg * 8);        // K = k + hg*8 .. +7
        af.h[1] = *(const u16x8*)(arow + k + hg * 8 + 16);   // K = +16 .. +23
        FragU bf;
        bf.h[0] = *(const u16x8*)(brow + k + hg * 8);        // K = k + hg*8 .. +7
        bf.h[1] = *(const u16x8*)(brow + k + hg * 8 + 16);   // K = +16 .. +23
        u16x16 bfrag = bf.v;
        __builtin_prefetch(brow + k + 256, 0, 0);
        acc = wmma_bf16(af.v, bfrag, acc);
    }

    float b = bias ? bias[tileN + l16] : 0.f;
    float* crow = C + (size_t)(tileM + hg * 8) * DI + tileN + l16;
#pragma unroll
    for (int r = 0; r < 8; ++r)
        crow[(size_t)r * DI] = acc[r] + b;
}

// ---------------------------------------------------------------- delta block
// out = x + beta * k_hat * (v - k_hat . x), per row; one block per row.
__global__ __launch_bounds__(256) void k_delta_row(const float* __restrict__ x,
                                                   const float* __restrict__ kraw,
                                                   const float* __restrict__ vin,
                                                   const float* __restrict__ dbw,   // D (offset i*D)
                                                   const float* __restrict__ dWv,   // D (offset i*D)
                                                   const float* __restrict__ dbb4,  // 4
                                                   const float* __restrict__ dbv4,  // 4
                                                   int i,
                                                   float* __restrict__ outF,        // nullable
                                                   u16* __restrict__ outB) {        // nullable
    const int row = blockIdx.x;
    const int t   = threadIdx.x;
    const float* xr = x    + (size_t)row * DI;
    const float* kr = kraw + (size_t)row * DI;
    const float* vr = vin  + (size_t)row * DI;

    const int j = t * 4;   // D=1024, 256 threads * 4
    float4 kx = *(const float4*)(kr + j);
    float4 xx = *(const float4*)(xr + j);
    float4 vv = *(const float4*)(vr + j);
    float4 bw = *(const float4*)(dbw + j);
    float4 wv = *(const float4*)(dWv + j);

    float s_n = kx.x*kx.x + kx.y*kx.y + kx.z*kx.z + kx.w*kx.w;
    float s_k = kx.x*xx.x + kx.y*xx.y + kx.z*xx.z + kx.w*xx.w;
    float s_b = xx.x*bw.x + xx.y*bw.y + xx.z*bw.z + xx.w*bw.w;
    float s_v = vv.x*wv.x + vv.y*wv.y + vv.z*wv.z + vv.w*wv.w;
#pragma unroll
    for (int m = 16; m; m >>= 1) {
        s_n += __shfl_xor(s_n, m, 32);
        s_k += __shfl_xor(s_k, m, 32);
        s_b += __shfl_xor(s_b, m, 32);
        s_v += __shfl_xor(s_v, m, 32);
    }
    __shared__ float red[8][4];
    if ((t & 31) == 0) {
        red[t >> 5][0] = s_n; red[t >> 5][1] = s_k;
        red[t >> 5][2] = s_b; red[t >> 5][3] = s_v;
    }
    __syncthreads();
    float nrm = 0.f, kh = 0.f, bwd = 0.f, vvd = 0.f;
#pragma unroll
    for (int w = 0; w < 8; ++w) {
        nrm += red[w][0]; kh += red[w][1]; bwd += red[w][2]; vvd += red[w][3];
    }
    const float denom = sqrtf(nrm) + 1e-8f;
    const float kdoth = kh / denom;
    const float beta  = 2.f / (1.f + __expf(-(bwd + dbb4[i])));
    const float v     = vvd + dbv4[i];
    const float coef  = beta * (v - kdoth) / denom;

    float4 o;
    o.x = xx.x + coef * kx.x; o.y = xx.y + coef * kx.y;
    o.z = xx.z + coef * kx.z; o.w = xx.w + coef * kx.w;
    if (outF) *(float4*)(outF + (size_t)row * DI + j) = o;
    if (outB) {
        u16x4 ob;
        ob[0] = f2bf(o.x); ob[1] = f2bf(o.y); ob[2] = f2bf(o.z); ob[3] = f2bf(o.w);
        *(u16x4*)(outB + (size_t)row * DI + j) = ob;
    }
}

// ---------------------------------------------------------------- flash attention
// One wave per 16-query tile of one (b,h); 32 keys per iteration; bf16 WMMA.
__global__ __launch_bounds__(128) void k_flash_attn(const u16* __restrict__ qf,
                                                    const u16* __restrict__ kf,
                                                    const u16* __restrict__ vf,
                                                    float* __restrict__ O) {
    const int lane = threadIdx.x & 31;
    const int wave = threadIdx.x >> 5;
    const int hg   = lane >> 4;
    const int l16  = lane & 15;

    const int tile = blockIdx.x * 4 + wave;        // 0..4095
    const int qt = tile & (SS / 16 - 1);           // query tile in sequence
    const int bh = tile >> 7;                      // 0..31
    const int b  = bh >> 4;
    const int h  = bh & 15;
    const size_t base = (size_t)b * SS * DI + (size_t)h * HDIM;
    const int q0 = qt * 16;

    __shared__ u16 ldsP[4][16 * 32] __attribute__((aligned(32)));   // probs tile
    __shared__ u16 ldsV[4][HDIM * 32] __attribute__((aligned(32))); // V^T [d][key]
    u16* Pl = ldsP[wave];
    u16* Vt = ldsV[wave];

    // Q A-fragments (16x32 each over d=0..31 and d=32..63), kept in registers
    const u16* qrow = qf + base + (size_t)(q0 + l16) * DI;
    FragU qa0, qa1;
    qa0.h[0] = *(const u16x8*)(qrow + hg * 8);
    qa0.h[1] = *(const u16x8*)(qrow + hg * 8 + 16);
    qa1.h[0] = *(const u16x8*)(qrow + 32 + hg * 8);
    qa1.h[1] = *(const u16x8*)(qrow + 32 + hg * 8 + 16);

    float mrow[8], lrow[8];
    v8f o0 = {0.f,0.f,0.f,0.f,0.f,0.f,0.f,0.f};
    v8f o1 = o0, o2 = o0, o3 = o0;
#pragma unroll
    for (int r = 0; r < 8; ++r) { mrow[r] = -1e30f; lrow[r] = 0.f; }

    const float sc = 0.125f;  // 1/sqrt(64)

    for (int kb = 0; kb < SS; kb += 32) {
        // ---- scores S = Q K^T for 32 keys: two 16x16 C tiles
        const u16* krow0 = kf + base + (size_t)(kb + l16) * DI;
        const u16* krow1 = kf + base + (size_t)(kb + 16 + l16) * DI;
        FragU f00, f01, f10, f11;
        f00.h[0] = *(const u16x8*)(krow0 + hg * 8);      f00.h[1] = *(const u16x8*)(krow0 + hg * 8 + 16);
        f01.h[0] = *(const u16x8*)(krow0 + 32 + hg * 8); f01.h[1] = *(const u16x8*)(krow0 + 32 + hg * 8 + 16);
        f10.h[0] = *(const u16x8*)(krow1 + hg * 8);      f10.h[1] = *(const u16x8*)(krow1 + hg * 8 + 16);
        f11.h[0] = *(const u16x8*)(krow1 + 32 + hg * 8); f11.h[1] = *(const u16x8*)(krow1 + 32 + hg * 8 + 16);
        u16x16 kb00 = f00.v;
        u16x16 kb01 = f01.v;
        u16x16 kb10 = f10.v;
        u16x16 kb11 = f11.v;
        __builtin_prefetch(krow0 + 32 * DI, 0, 0);

        v8f s0 = {0.f,0.f,0.f,0.f,0.f,0.f,0.f,0.f};
        v8f s1 = s0;
        s0 = wmma_bf16(qa0.v, kb00, s0);
        s0 = wmma_bf16(qa1.v, kb01, s0);
        s1 = wmma_bf16(qa0.v, kb10, s1);
        s1 = wmma_bf16(qa1.v, kb11, s1);

        // ---- stage V tile transposed into LDS: Vt[d][key], key = lane
        const u16* vrow = vf + base + (size_t)(kb + lane) * DI;
#pragma unroll
        for (int d = 0; d < HDIM; d += 8) {
            u16x8 vvv = *(const u16x8*)(vrow + d);
#pragma unroll
            for (int e = 0; e < 8; ++e) Vt[(d + e) * 32 + lane] = vvv[e];
        }

        // ---- online softmax (row groups == 16-lane halves of C layout)
#pragma unroll
        for (int r = 0; r < 8; ++r) {
            float a = s0[r] * sc, c = s1[r] * sc;
            float mx = fmaxf(a, c);
            mx = fmaxf(mx, __shfl_xor(mx, 1, 32));
            mx = fmaxf(mx, __shfl_xor(mx, 2, 32));
            mx = fmaxf(mx, __shfl_xor(mx, 4, 32));
            mx = fmaxf(mx, __shfl_xor(mx, 8, 32));
            float nm = fmaxf(mrow[r], mx);
            float ea = __expf(a - nm), ec = __expf(c - nm);
            float scale = __expf(mrow[r] - nm);
            float rs = ea + ec;
            rs += __shfl_xor(rs, 1, 32);
            rs += __shfl_xor(rs, 2, 32);
            rs += __shfl_xor(rs, 4, 32);
            rs += __shfl_xor(rs, 8, 32);
            lrow[r] = lrow[r] * scale + rs;
            mrow[r] = nm;
            o0[r] *= scale; o1[r] *= scale; o2[r] *= scale; o3[r] *= scale;
            Pl[(r + 8 * hg) * 32 + l16]      = f2bf(ea);
            Pl[(r + 8 * hg) * 32 + 16 + l16] = f2bf(ec);
        }
        asm volatile("s_wait_dscnt 0" ::: "memory");

        // ---- reload P as 16x32 A-fragment
        FragU pa;
        const u16* prow = Pl + l16 * 32;
        pa.h[0] = *(const u16x8*)(prow + hg * 8);
        pa.h[1] = *(const u16x8*)(prow + hg * 8 + 16);

        // ---- O += P * V : four 16x16 C tiles over d groups
#pragma unroll
        for (int g = 0; g < 4; ++g) {
            FragU vf2;
            vf2.h[0] = *(const u16x8*)(Vt + (g * 16 + l16) * 32 + hg * 8);
            vf2.h[1] = *(const u16x8*)(Vt + (g * 16 + l16) * 32 + hg * 8 + 16);
            u16x16 vb = vf2.v;
            v8f* op = (g == 0) ? &o0 : (g == 1) ? &o1 : (g == 2) ? &o2 : &o3;
            *op = wmma_bf16(pa.v, vb, *op);
        }
        asm volatile("s_wait_dscnt 0" ::: "memory");
    }

    // ---- normalize and store
#pragma unroll
    for (int r = 0; r < 8; ++r) {
        float inv = 1.f / lrow[r];
        o0[r] *= inv; o1[r] *= inv; o2[r] *= inv; o3[r] *= inv;
    }
    float* obase = O + base + (size_t)(q0 + 8 * hg) * DI + l16;
#pragma unroll
    for (int r = 0; r < 8; ++r) {
        float* orow = obase + (size_t)r * DI;
        orow[0]  = o0[r];
        orow[16] = o1[r];
        orow[32] = o2[r];
        orow[48] = o3[r];
    }
}

// ---------------------------------------------------------------- layernorm
__global__ __launch_bounds__(256) void k_layernorm(const float* __restrict__ y,
                                                   const float* __restrict__ g,
                                                   const float* __restrict__ bta,
                                                   float* __restrict__ out) {
    const int row = blockIdx.x;
    const int t   = threadIdx.x;
    const float* yr = y + (size_t)row * DI;
    float4 v = *(const float4*)(yr + t * 4);
    float s  = v.x + v.y + v.z + v.w;
    float s2 = v.x*v.x + v.y*v.y + v.z*v.z + v.w*v.w;
#pragma unroll
    for (int m = 16; m; m >>= 1) {
        s  += __shfl_xor(s,  m, 32);
        s2 += __shfl_xor(s2, m, 32);
    }
    __shared__ float red[8][2];
    if ((t & 31) == 0) { red[t >> 5][0] = s; red[t >> 5][1] = s2; }
    __syncthreads();
    float S = 0.f, S2 = 0.f;
#pragma unroll
    for (int w = 0; w < 8; ++w) { S += red[w][0]; S2 += red[w][1]; }
    const float mu  = S * (1.f / DI);
    const float var = S2 * (1.f / DI) - mu * mu;
    const float rstd = rsqrtf(var + 1e-5f);
    float4 gg = *(const float4*)(g   + t * 4);
    float4 bb = *(const float4*)(bta + t * 4);
    float4 o;
    o.x = (v.x - mu) * rstd * gg.x + bb.x;
    o.y = (v.y - mu) * rstd * gg.y + bb.y;
    o.z = (v.z - mu) * rstd * gg.z + bb.z;
    o.w = (v.w - mu) * rstd * gg.w + bb.w;
    *(float4*)(out + (size_t)row * DI + t * 4) = o;
}

// ---------------------------------------------------------------- launch
extern "C" void kernel_launch(void* const* d_in, const int* in_sizes, int n_in,
                              void* d_out, int out_size, void* d_ws, size_t ws_size,
                              hipStream_t stream) {
    const float* x   = (const float*)d_in[0];
    const float* Wq  = (const float*)d_in[1];
    const float* bq  = (const float*)d_in[2];
    const float* Wk  = (const float*)d_in[3];
    const float* bk  = (const float*)d_in[4];
    const float* Wv  = (const float*)d_in[5];
    const float* bv  = (const float*)d_in[6];
    const float* Wo  = (const float*)d_in[7];
    const float* bo  = (const float*)d_in[8];
    const float* dWk = (const float*)d_in[9];    // (4,D,D)
    const float* dbw = (const float*)d_in[10];   // (4,D)
    const float* dbb = (const float*)d_in[11];   // (4,)
    const float* dWv = (const float*)d_in[12];   // (4,D)
    const float* dbv = (const float*)d_in[13];   // (4,)
    const float* lng = (const float*)d_in[14];
    const float* lnb = (const float*)d_in[15];
    float* out = (float*)d_out;

    char* ws = (char*)d_ws;
    const size_t MB = 1u << 20;
    u16*   xb   = (u16*)(ws + 0 * MB);           // 8 MiB  bf16 x
    u16*   wt   = (u16*)(ws + 8 * MB);           // 16 MiB: [Wq,Wk,Wv,Wo,dWk0..3]^T bf16
    float* t0   = (float*)(ws + 24 * MB);        // 16 MiB f32 scratch (v_in proj)
    float* t1   = (float*)(ws + 40 * MB);        // 16 MiB f32 scratch (k_raw)
    u16*   qfb  = (u16*)(ws + 56 * MB);          // 8 MiB
    u16*   kfb  = (u16*)(ws + 64 * MB);          // 8 MiB
    u16*   vfb  = (u16*)(ws + 72 * MB);          // 8 MiB
    float* Obuf = (float*)(ws + 80 * MB);        // 16 MiB attention output f32
    u16*   Ob   = (u16*)(ws + 96 * MB);          // 8 MiB attention output bf16
    float* ybuf = (float*)(ws + 104 * MB);       // 16 MiB  (total 120 MiB)

    const size_t DD = (size_t)DI * DI;
    const int packBlocks = MROWS * DI / (256 * 4);   // 4096
    const dim3 gT(DD / 256);                         // 4096
    const dim3 gg(DI / 64, MROWS / 16);              // (16,256)

    // Stage inputs as bf16
    k_pack_bf16<<<packBlocks, 256, 0, stream>>>(x, xb);
    k_packT_bf16<<<gT, 256, 0, stream>>>(Wq, wt + 0 * DD);
    k_packT_bf16<<<gT, 256, 0, stream>>>(Wk, wt + 1 * DD);
    k_packT_bf16<<<gT, 256, 0, stream>>>(Wv, wt + 2 * DD);
    k_packT_bf16<<<gT, 256, 0, stream>>>(Wo, wt + 3 * DD);
    k_packT_bf16<<<gT, 256, 0, stream>>>(dWk + 0 * DD, wt + 4 * DD);
    k_packT_bf16<<<gT, 256, 0, stream>>>(dWk + 1 * DD, wt + 5 * DD);
    k_packT_bf16<<<gT, 256, 0, stream>>>(dWk + 2 * DD, wt + 6 * DD);
    k_packT_bf16<<<gT, 256, 0, stream>>>(dWk + 3 * DD, wt + 7 * DD);

    // q = delta(x, x@Wq+bq, 0) -> qfb (bf16)
    k_gemm_bf16<<<gg, 128, 0, stream>>>(xb, wt + 0 * DD, bq, t0);
    k_gemm_bf16<<<gg, 128, 0, stream>>>(xb, wt + 4 * DD, nullptr, t1);
    k_delta_row<<<MROWS, 256, 0, stream>>>(x, t1, t0, dbw + 0 * DI, dWv + 0 * DI,
                                           dbb, dbv, 0, nullptr, qfb);
    // k = delta(x, x@Wk+bk, 1) -> kfb
    k_gemm_bf16<<<gg, 128, 0, stream>>>(xb, wt + 1 * DD, bk, t0);
    k_gemm_bf16<<<gg, 128, 0, stream>>>(xb, wt + 5 * DD, nullptr, t1);
    k_delta_row<<<MROWS, 256, 0, stream>>>(x, t1, t0, dbw + 1 * DI, dWv + 1 * DI,
                                           dbb, dbv, 1, nullptr, kfb);
    // v = delta(x, x@Wv+bv, 2) -> vfb
    k_gemm_bf16<<<gg, 128, 0, stream>>>(xb, wt + 2 * DD, bv, t0);
    k_gemm_bf16<<<gg, 128, 0, stream>>>(xb, wt + 6 * DD, nullptr, t1);
    k_delta_row<<<MROWS, 256, 0, stream>>>(x, t1, t0, dbw + 2 * DI, dWv + 2 * DI,
                                           dbb, dbv, 2, nullptr, vfb);

    // attention: 4096 query tiles / 4 waves per block
    k_flash_attn<<<1024, 128, 0, stream>>>(qfb, kfb, vfb, Obuf);

    // out = attn_out @ Wo + bo
    k_pack_bf16<<<packBlocks, 256, 0, stream>>>(Obuf, Ob);
    k_gemm_bf16<<<gg, 128, 0, stream>>>(Ob, wt + 3 * DD, bo, t0);

    // y = delta(x, out, 3) -> ybuf (f32)
    k_gemm_bf16<<<gg, 128, 0, stream>>>(xb, wt + 7 * DD, nullptr, t1);
    k_delta_row<<<MROWS, 256, 0, stream>>>(x, t1, t0, dbw + 3 * DI, dWv + 3 * DI,
                                           dbb, dbv, 3, ybuf, nullptr);

    // LayerNorm -> d_out
    k_layernorm<<<MROWS, 256, 0, stream>>>(ybuf, lng, lnb, out);
}
